// FusedDeepSeekMoEMLP_21861383536938
// MI455X (gfx1250) — compile-verified, matrix operand order checked by reading, not run on hardware
//
#include <hip/hip_runtime.h>

typedef __attribute__((ext_vector_type(16))) __bf16 v16bf;
typedef __attribute__((ext_vector_type(8)))  float  v8f;

// Problem constants (match reference)
constexpr int BB = 2, T = 1024, DIM = 768, E = 4;
constexpr int HDIM = 4 * DIM;          // 3072
constexpr int N = BB * T;              // 2048 tokens
constexpr int CAP = 2064;              // per-expert bucket capacity (>= 2048)
constexpr int TILE_M = 64;             // token rows per workgroup tile
constexpr int CHUNK = 768;             // HDIM chunk held in LDS
constexpr int NCHUNK = HDIM / CHUNK;   // 4
constexpr int MAX_TILES = 68;          // ceil((4096 + 4*63)/64)

// ---- workspace layout (bytes) ----
constexpr size_t WS_COUNTS   = 0;                                   // E ints
constexpr size_t WS_ZSUM     = 16;                                  // float
constexpr size_t WS_PROBSUM  = 32;                                  // E floats
constexpr size_t WS_NTILES   = 48;                                  // int
constexpr size_t WS_TILE_E   = 64;                                  // MAX_TILES ints
constexpr size_t WS_TILE_OFF = WS_TILE_E + (size_t)MAX_TILES * 4;   // MAX_TILES ints
constexpr size_t WS_BTOK     = ((WS_TILE_OFF + (size_t)MAX_TILES * 4 + 255) / 256) * 256;
constexpr size_t WS_BW       = WS_BTOK + (size_t)E * CAP * 4;
constexpr size_t WS_WUP_T    = ((WS_BW + (size_t)E * CAP * 4 + 255) / 256) * 256; // bf16 [E][HDIM][DIM]
constexpr size_t WS_WDN      = WS_WUP_T + (size_t)E * HDIM * DIM * 2;             // bf16 [E][DIM][HDIM]
constexpr size_t WS_XBF      = WS_WDN + (size_t)E * DIM * HDIM * 2;               // bf16 [N][DIM]

// LDS layout (byte offsets inside dynamic smem; dynamic LDS starts at offset 0)
constexpr int LDS_TOK = 0;                       // 64 ints
constexpr int LDS_W   = 256;                     // 64 floats
constexpr int LDS_X   = 512;                     // 64 x 768 bf16 = 96 KB
constexpr int LDS_H   = 512 + TILE_M * DIM * 2;  // 64 x 768 bf16 = 96 KB
constexpr int LDS_BYTES = LDS_H + TILE_M * CHUNK * 2; // 197,120 B (< 320 KB WGP LDS)

__device__ __forceinline__ unsigned short f2bf(float f) {
    unsigned u = __float_as_uint(f);
    u += 0x7FFFu + ((u >> 16) & 1u);          // round-to-nearest-even
    return (unsigned short)(u >> 16);
}

union Frag { uint4 q[2]; v16bf v; };

// ---------------- kernel 0: zero output + meta ----------------
__global__ void moe_zero(float* __restrict__ out, char* __restrict__ ws) {
    size_t i = (size_t)blockIdx.x * blockDim.x + threadIdx.x;
    if (i < (size_t)N * DIM) out[i] = 0.0f;
    if (blockIdx.x == 0 && threadIdx.x < 16) ((unsigned*)ws)[threadIdx.x] = 0u;
}

// ---------------- kernel 1: router (softmax, top-2, bucket append, loss accum) ----------------
__global__ void moe_router(const float* __restrict__ x, const float* __restrict__ rw,
                           char* __restrict__ ws) {
    int n = blockIdx.x * blockDim.x + threadIdx.x;
    if (n >= N) return;
    const float4* xv = (const float4*)(x + (size_t)n * DIM);
    float acc[E] = {0.f, 0.f, 0.f, 0.f};
    for (int i = 0; i < DIM / 4; ++i) {
        float4 xe = xv[i];
#pragma unroll
        for (int e = 0; e < E; ++e) {
            float4 we = ((const float4*)(rw + (size_t)e * DIM))[i];
            acc[e] += xe.x * we.x + xe.y * we.y + xe.z * we.z + xe.w * we.w;
        }
    }
    float m = acc[0];
#pragma unroll
    for (int e = 1; e < E; ++e) m = fmaxf(m, acc[e]);
    float p[E]; float s = 0.f;
#pragma unroll
    for (int e = 0; e < E; ++e) { p[e] = __expf(acc[e] - m); s += p[e]; }
    float inv = 1.0f / s;
#pragma unroll
    for (int e = 0; e < E; ++e) p[e] *= inv;
    float lse = m + __logf(s);
    atomicAdd((float*)(ws + WS_ZSUM), lse * lse);
#pragma unroll
    for (int e = 0; e < E; ++e) atomicAdd((float*)(ws + WS_PROBSUM) + e, p[e]);

    int i0 = 0; float p0 = p[0];
#pragma unroll
    for (int e = 1; e < E; ++e) if (p[e] > p0) { p0 = p[e]; i0 = e; }
    int i1 = (i0 == 0) ? 1 : 0; float p1 = p[i1];
#pragma unroll
    for (int e = 0; e < E; ++e) if (e != i0 && p[e] > p1) { p1 = p[e]; i1 = e; }
    float wsum = p0 + p1;
    int*   counts = (int*)(ws + WS_COUNTS);
    int*   btok   = (int*)(ws + WS_BTOK);
    float* bw     = (float*)(ws + WS_BW);
    int pos0 = atomicAdd(&counts[i0], 1);
    btok[i0 * CAP + pos0] = n; bw[i0 * CAP + pos0] = p0 / wsum;
    int pos1 = atomicAdd(&counts[i1], 1);
    btok[i1 * CAP + pos1] = n; bw[i1 * CAP + pos1] = p1 / wsum;
}

// ---------------- kernel 2: scheduler (pad buckets to 64, emit 64-row tile descriptors) ----------------
__global__ void moe_sched(char* __restrict__ ws) {
    if (threadIdx.x != 0 || blockIdx.x != 0) return;
    int*   counts = (int*)(ws + WS_COUNTS);
    int*   te     = (int*)(ws + WS_TILE_E);
    int*   toff   = (int*)(ws + WS_TILE_OFF);
    int*   btok   = (int*)(ws + WS_BTOK);
    float* bw     = (float*)(ws + WS_BW);
    int tot = 0;
    for (int e = 0; e < E; ++e) {
        int c = counts[e];
        int padded = (c + 63) & ~63;
        for (int i = c; i < padded; ++i) { btok[e * CAP + i] = 0; bw[e * CAP + i] = 0.0f; }
        for (int t = 0; t < padded / TILE_M; ++t) { te[tot] = e; toff[tot] = t * TILE_M; ++tot; }
    }
    *(int*)(ws + WS_NTILES) = tot;
}

// ---------------- kernel 3a: combine shared+routed weights -> bf16 ----------------
__global__ void moe_prep_w(const float* __restrict__ su, const float* __restrict__ sd,
                           const float* __restrict__ ru, const float* __restrict__ rd,
                           char* __restrict__ ws) {
    size_t idx = (size_t)blockIdx.x * blockDim.x + threadIdx.x;
    if (idx >= (size_t)E * DIM * HDIM) return;
    int e = (int)(idx / ((size_t)DIM * HDIM));
    int r = (int)(idx % ((size_t)DIM * HDIM));
    int d = r / HDIM, h = r % HDIM;
    unsigned short* wup = (unsigned short*)(ws + WS_WUP_T);
    unsigned short* wdn = (unsigned short*)(ws + WS_WDN);
    wdn[idx] = f2bf(sd[r] + rd[idx]);                                 // [e][d][h]
    wup[((size_t)e * HDIM + h) * DIM + d] = f2bf(su[r] + ru[idx]);    // transposed [e][h][d]
}

// ---------------- kernel 3b: x -> bf16 ----------------
__global__ void moe_prep_x(const float* __restrict__ x, char* __restrict__ ws) {
    size_t i = (size_t)blockIdx.x * blockDim.x + threadIdx.x;
    if (i < (size_t)N * DIM) ((unsigned short*)(ws + WS_XBF))[i] = f2bf(x[i]);
}

// ---------------- kernel 4: fused 64-token tile: up -> relu^2 -> down (chunked) -> scatter ----------------
__global__ __launch_bounds__(512) void moe_gemm(float* __restrict__ out,
                                                char* __restrict__ ws) {
    extern __shared__ char smem[];
    int*            s_tok = (int*)(smem + LDS_TOK);             // 64 ints
    float*          s_w   = (float*)(smem + LDS_W);             // 64 floats
    unsigned short* s_x   = (unsigned short*)(smem + LDS_X);    // 64 x 768 bf16 (resident)
    unsigned short* s_h   = (unsigned short*)(smem + LDS_H);    // 64 x 768 bf16 (per chunk)

    int tile = blockIdx.x;
    int ntiles = *(const int*)(ws + WS_NTILES);
    if (tile >= ntiles) return;
    int e   = ((const int*)(ws + WS_TILE_E))[tile];
    int off = ((const int*)(ws + WS_TILE_OFF))[tile];
    int tid = threadIdx.x;
    if (tid < TILE_M) {
        s_tok[tid] = ((const int*)(ws + WS_BTOK))[e * CAP + off + tid];
        s_w[tid]   = ((const float*)(ws + WS_BW))[e * CAP + off + tid];
    }
    __syncthreads();

    // ---- async gather: 64 token rows of bf16 x -> LDS (96 KB = 6144 x 16B, 12 per thread) ----
    {
        unsigned long long xb = (unsigned long long)(uintptr_t)(ws + WS_XBF);
#pragma unroll
        for (int p = 0; p < (TILE_M * DIM * 2 / 16) / 512; ++p) {
            int i = tid + p * 512;
            int row  = i / (DIM * 2 / 16);                 // 96 chunks per row
            int colb = (i - row * (DIM * 2 / 16)) * 16;
            unsigned long long ga = xb + (unsigned long long)s_tok[row] * (DIM * 2) + colb;
            unsigned la = (unsigned)(LDS_X + row * (DIM * 2) + colb);
            asm volatile("global_load_async_to_lds_b128 %0, %1, off"
                         :: "v"(la), "v"(ga) : "memory");
        }
        asm volatile("s_wait_asynccnt 0x0" ::: "memory");
    }
    __syncthreads();

    int lane = tid & 31, wave = tid >> 5;   // 16 waves
    int rn  = lane & 15;                    // A-row / B-col / D-col for this lane
    int kb  = (lane >> 4) << 4;             // K half: 0 or 16
    int mhi = (lane >> 4) * 8;              // D-row base for this lane half
    const unsigned short* wup = (const unsigned short*)(ws + WS_WUP_T) + (size_t)e * HDIM * DIM;
    const unsigned short* wdn = (const unsigned short*)(ws + WS_WDN)   + (size_t)e * DIM * HDIM;

    v8f accD[12] = {};   // persistent down accumulators: 3 d-tiles x 4 m-tiles

    for (int c = 0; c < NCHUNK; ++c) {
        // ---- UP: h_chunk[64 x 768] = x_tile @ Wup[:, c*768 : (c+1)*768], relu^2 -> LDS ----
#pragma unroll 1
        for (int ntl = 0; ntl < CHUNK / 16 / 16; ++ntl) {   // 3 n-tiles per wave per chunk
            int nt  = wave + ntl * 16;                      // 0..47 within chunk
            int ntg = c * (CHUNK / 16) + nt;                // global h column tile
            v8f accU[4] = {};
            const unsigned short* bbase = wup + (size_t)(ntg * 16 + rn) * DIM + kb;
#pragma unroll 2
            for (int kt = 0; kt < DIM / 32; ++kt) {
                __builtin_prefetch(bbase + kt * 32 + 512, 0, 1);
                Frag b;
                const uint4* bp = (const uint4*)(bbase + kt * 32);
                b.q[0] = bp[0]; b.q[1] = bp[1];
#pragma unroll
                for (int mt = 0; mt < 4; ++mt) {            // reuse B across 4 row sub-tiles
                    Frag a;
                    const uint4* ap = (const uint4*)(s_x + (mt * 16 + rn) * DIM + kt * 32 + kb);
                    a.q[0] = ap[0]; a.q[1] = ap[1];
                    accU[mt] = __builtin_amdgcn_wmma_f32_16x16x32_bf16(
                        false, a.v, false, b.v, (short)0, accU[mt], false, false);
                }
            }
#pragma unroll
            for (int mt = 0; mt < 4; ++mt)
#pragma unroll
                for (int r = 0; r < 8; ++r) {
                    float v = accU[mt][r];
                    v = v > 0.0f ? v * v : 0.0f;            // relu(h)^2
                    s_h[(mt * 16 + mhi + r) * CHUNK + nt * 16 + rn] = f2bf(v);
                }
        }
        __syncthreads();

        // ---- DOWN: accD += h_chunk @ Wdn[:, c*768:(c+1)*768]^T ----
#pragma unroll 1
        for (int kt = 0; kt < CHUNK / 32; ++kt) {
            Frag a[4];
#pragma unroll
            for (int mt = 0; mt < 4; ++mt) {
                const uint4* ap = (const uint4*)(s_h + (mt * 16 + rn) * CHUNK + kt * 32 + kb);
                a[mt].q[0] = ap[0]; a[mt].q[1] = ap[1];
            }
#pragma unroll
            for (int dti = 0; dti < 3; ++dti) {
                int dt = wave + dti * 16;                   // 0..47 output d-tile
                const unsigned short* bbase =
                    wdn + (size_t)(dt * 16 + rn) * HDIM + c * CHUNK + kt * 32 + kb;
                __builtin_prefetch(bbase + 512, 0, 1);
                Frag b;
                const uint4* bp = (const uint4*)bbase;
                b.q[0] = bp[0]; b.q[1] = bp[1];
#pragma unroll
                for (int mt = 0; mt < 4; ++mt)
                    accD[dti * 4 + mt] = __builtin_amdgcn_wmma_f32_16x16x32_bf16(
                        false, a[mt].v, false, b.v, (short)0, accD[dti * 4 + mt], false, false);
            }
        }
        __syncthreads();    // before next chunk overwrites s_h
    }

    // ---- weighted atomic scatter ----
#pragma unroll
    for (int dti = 0; dti < 3; ++dti) {
        int dt = wave + dti * 16;
#pragma unroll
        for (int mt = 0; mt < 4; ++mt)
#pragma unroll
            for (int r = 0; r < 8; ++r) {
                int m = mt * 16 + mhi + r;
                float val = accD[dti * 4 + mt][r] * s_w[m];
                atomicAdd(out + (size_t)s_tok[m] * DIM + dt * 16 + rn, val);
            }
    }
}

// ---------------- kernel 5: finalize aux losses ----------------
__global__ void moe_final(float* __restrict__ out, const char* __restrict__ ws) {
    if (threadIdx.x != 0 || blockIdx.x != 0) return;
    const int*   counts = (const int*)(ws + WS_COUNTS);
    const float* ps     = (const float*)(ws + WS_PROBSUM);
    float z = *(const float*)(ws + WS_ZSUM) / (float)N;
    float lb = 0.0f;
    for (int e = 0; e < E; ++e)
        lb += ((float)counts[e] / (float)N) * (ps[e] / (float)N);
    lb *= (float)E;
    out[(size_t)N * DIM + 0] = lb;
    out[(size_t)N * DIM + 1] = z;
}

extern "C" void kernel_launch(void* const* d_in, const int* in_sizes, int n_in,
                              void* d_out, int out_size, void* d_ws, size_t ws_size,
                              hipStream_t stream) {
    (void)in_sizes; (void)n_in; (void)out_size; (void)ws_size;
    const float* x  = (const float*)d_in[0];
    const float* rw = (const float*)d_in[1];
    const float* su = (const float*)d_in[2];
    const float* sd = (const float*)d_in[3];
    const float* ru = (const float*)d_in[4];
    const float* rd = (const float*)d_in[5];
    float* out = (float*)d_out;
    char*  ws  = (char*)d_ws;

    hipFuncSetAttribute((const void*)moe_gemm,
                        hipFuncAttributeMaxDynamicSharedMemorySize, LDS_BYTES);

    moe_zero  <<<(N * DIM + 255) / 256, 256, 0, stream>>>(out, ws);
    moe_router<<<(N + 255) / 256, 256, 0, stream>>>(x, rw, ws);
    moe_sched <<<1, 32, 0, stream>>>(ws);
    size_t wtot = (size_t)E * DIM * HDIM;
    moe_prep_w<<<(unsigned)((wtot + 255) / 256), 256, 0, stream>>>(su, sd, ru, rd, ws);
    moe_prep_x<<<(N * DIM + 255) / 256, 256, 0, stream>>>(x, ws);
    moe_gemm  <<<MAX_TILES, 512, LDS_BYTES, stream>>>(out, ws);
    moe_final <<<1, 32, 0, stream>>>(out, ws);
}
